// GAT_15968688407067
// MI455X (gfx1250) — matrix operand order repaired and checked by reading, not compiled, not run on hardware
//
#include <hip/hip_runtime.h>
#include <hip/hip_bf16.h>
#include <math.h>

typedef __attribute__((ext_vector_type(16))) _Float16 v16h;
typedef __attribute__((ext_vector_type(8)))  _Float16 v8h;
typedef __attribute__((ext_vector_type(8)))  float    v8f;

#define NEG_SLOPE 0.2f

// ---------------------------------------------------------------------------
// f32 -> f16 conversion (activations)
// ---------------------------------------------------------------------------
__global__ __launch_bounds__(256)
void cvt_f32_f16(const float* __restrict__ in, _Float16* __restrict__ out, size_t n) {
  size_t i = (size_t)blockIdx.x * blockDim.x + threadIdx.x;
  if (i < n) out[i] = (_Float16)in[i];
}

// ---------------------------------------------------------------------------
// Pack W[K,M] (f32 row-major) into WMMA B-fragment order (f16):
//   Bp[(((kt*CT + ct)*32 + lane)*16) + e] = W[(kt*32 + (lane>>4)*16 + e)*M + ct*16 + (lane&15)]
// so each lane's 16-half B fragment is one contiguous 32-byte run.
// ---------------------------------------------------------------------------
__global__ __launch_bounds__(256)
void pack_w16(const float* __restrict__ W, _Float16* __restrict__ Bp,
              int K, int M) {
  const int CT = M >> 4;
  size_t idx = (size_t)blockIdx.x * blockDim.x + threadIdx.x;
  size_t tot = (size_t)K * M;
  if (idx >= tot) return;
  int e    = (int)(idx & 15);
  int lane = (int)((idx >> 4) & 31);
  int rest = (int)(idx >> 9);
  int ct   = rest % CT;
  int kt   = rest / CT;
  int g = lane >> 4, n = lane & 15;
  int k   = kt * 32 + g * 16 + e;
  int col = ct * 16 + n;
  Bp[idx] = (_Float16)W[(size_t)k * M + col];
}

// ---------------------------------------------------------------------------
// WMMA GEMM:  C[N,M] = A16[N,K] * Bpacked + bias[M]   (f16 in, f32 out)
// block = 128 threads = 4 waves; each wave computes one 16x16 C tile.
// grid = (ceil(N/16), M/64). K multiple of 32, M multiple of 64.
// A fragment  = two contiguous b128 loads per lane.
// B fragment  = two contiguous b128 loads per lane (pre-packed order).
// ---------------------------------------------------------------------------
__global__ __launch_bounds__(128)
void gemm_wmma_bias(const _Float16* __restrict__ A, const _Float16* __restrict__ Bp,
                    const float* __restrict__ bias, float* __restrict__ C,
                    int Nrows, int K, int M) {
  const int wave = threadIdx.x >> 5;
  const int lane = threadIdx.x & 31;
  const int g    = lane >> 4;        // lane half
  const int lo   = lane & 15;
  const int rowBase = blockIdx.x * 16;
  const int ct      = blockIdx.y * 4 + wave;      // 16-col tile index
  const int colBase = ct * 16;
  const int CT      = M >> 4;

  int row  = rowBase + lo;
  int rowc = row < Nrows ? row : (Nrows - 1);      // clamp for safe loads
  const _Float16* Arow = A + (size_t)rowc * K + g * 16;
  const _Float16* Bfr  = Bp + ((size_t)ct * 32 + lane) * 16;  // + kt*CT*512
  const size_t Bstep   = (size_t)CT * 512;                    // halves per k-tile
  const int col = colBase + lo;

  union { v16h v; v8h h[2]; } a, b;
  v8f c = {};
  const int KT = K >> 5;
  for (int kt = 0; kt < KT; ++kt) {
    const _Float16* ap = Arow + kt * 32;
    a.h[0] = *(const v8h*)(ap);
    a.h[1] = *(const v8h*)(ap + 8);
    const _Float16* bp = Bfr + (size_t)kt * Bstep;
    b.h[0] = *(const v8h*)(bp);
    b.h[1] = *(const v8h*)(bp + 8);
    // (neg_a, A, neg_b, B, c_mod, C, reuse_a, reuse_b)
    c = __builtin_amdgcn_wmma_f32_16x16x32_f16(false, a.v, false, b.v, (short)0, c,
                                               false, false);
  }

  const float bv = bias[col];
#pragma unroll
  for (int v = 0; v < 8; ++v) {
    // C layout: element v -> row M = v + 8*g, col N = lane%16
    const int r = rowBase + v + (g << 3);
    if (r < Nrows) C[(size_t)r * M + col] = c[v] + bv;
  }
}

// ---------------------------------------------------------------------------
// per-layer init: acc[n][j] = bias_out[j]; m = -inf; s = 0
// ---------------------------------------------------------------------------
__global__ __launch_bounds__(256)
void init_layer(float* __restrict__ m, float* __restrict__ s,
                float* __restrict__ acc, const float* __restrict__ bo,
                int Nn, int H, int HC) {
  size_t idx = (size_t)blockIdx.x * blockDim.x + threadIdx.x;
  size_t tot = (size_t)Nn * HC;
  if (idx < tot) acc[idx] = bo[idx % HC];
  if (idx < (size_t)Nn * H) { m[idx] = -INFINITY; s[idx] = 0.f; }
}

__device__ inline void atomicMaxF(float* addr, float v) {
  if (v >= 0.f) atomicMax((int*)addr, __float_as_int(v));
  else          atomicMin((unsigned int*)addr, __float_as_uint(v));
}

__device__ inline void edge_endpoints(int e, const int* __restrict__ ei, int E,
                                      int& s_, int& d_) {
  if (e < E) { s_ = ei[e]; d_ = ei[E + e]; }
  else       { s_ = e - E; d_ = e - E; }      // self-loops appended at the end
}

// ---------------------------------------------------------------------------
// edge scores: e[e,h] = sum_c att[h,c] * leakyrelu(xl[src][h,c] + xr[dst][h,c])
// also atomic-max into m[dst,h].  Vectorized float4 reads.
// ---------------------------------------------------------------------------
__global__ __launch_bounds__(256)
void edge_scores(const float* __restrict__ xl, const float* __restrict__ xr,
                 const int* __restrict__ ei, const float* __restrict__ att,
                 float* __restrict__ p, float* __restrict__ m,
                 int E, int Ntot, int H, int Cc) {
  size_t idx = (size_t)blockIdx.x * blockDim.x + threadIdx.x;
  if (idx >= (size_t)Ntot * H) return;
  int e = (int)(idx / H), h = (int)(idx - (size_t)e * H);
  int s_, d_; edge_endpoints(e, ei, E, s_, d_);
  const int HC = H * Cc;
  const float4* xls = (const float4*)(xl + (size_t)s_ * HC + h * Cc);
  const float4* xrd = (const float4*)(xr + (size_t)d_ * HC + h * Cc);
  const float4* at  = (const float4*)(att + h * Cc);
  float acc = 0.f;
  const int C4 = Cc >> 2;
  for (int c = 0; c < C4; ++c) {
    float4 l = xls[c], r = xrd[c], w = at[c];
    float t;
    t = l.x + r.x; t = t > 0.f ? t : NEG_SLOPE * t; acc = fmaf(w.x, t, acc);
    t = l.y + r.y; t = t > 0.f ? t : NEG_SLOPE * t; acc = fmaf(w.y, t, acc);
    t = l.z + r.z; t = t > 0.f ? t : NEG_SLOPE * t; acc = fmaf(w.z, t, acc);
    t = l.w + r.w; t = t > 0.f ? t : NEG_SLOPE * t; acc = fmaf(w.w, t, acc);
  }
  p[idx] = acc;
  atomicMaxF(&m[(size_t)d_ * H + h], acc);
}

// p = exp(e - m[dst]); s[dst] += p
__global__ __launch_bounds__(256)
void edge_exp(float* __restrict__ p, const float* __restrict__ m,
              float* __restrict__ s, const int* __restrict__ ei,
              int E, int Ntot, int H) {
  size_t idx = (size_t)blockIdx.x * blockDim.x + threadIdx.x;
  if (idx >= (size_t)Ntot * H) return;
  int e = (int)(idx / H), h = (int)(idx - (size_t)e * H);
  int s_, d_; edge_endpoints(e, ei, E, s_, d_);
  (void)s_;
  float pv = __expf(p[idx] - m[(size_t)d_ * H + h]);
  p[idx] = pv;
  atomicAdd(&s[(size_t)d_ * H + h], pv);
}

// alpha = p / s[dst]
__global__ __launch_bounds__(256)
void alpha_norm(float* __restrict__ p, const float* __restrict__ s,
                const int* __restrict__ ei, int E, int Ntot, int H) {
  size_t idx = (size_t)blockIdx.x * blockDim.x + threadIdx.x;
  if (idx >= (size_t)Ntot * H) return;
  int e = (int)(idx / H), h = (int)(idx - (size_t)e * H);
  int s_, d_; edge_endpoints(e, ei, E, s_, d_);
  (void)s_;
  p[idx] = p[idx] / s[(size_t)d_ * H + h];
}

// acc[dst][j..j+3] += alpha[e, j/C] * xl[src][j..j+3]   (float4 gather, 4 atomics)
__global__ __launch_bounds__(256)
void aggregate4(const float* __restrict__ p, const float* __restrict__ xl,
                const int* __restrict__ ei, float* __restrict__ acc,
                int E, int Ntot, int H, int Cc) {
  const int HC  = H * Cc;
  const int HC4 = HC >> 2;
  size_t idx = (size_t)blockIdx.x * blockDim.x + threadIdx.x;
  if (idx >= (size_t)Ntot * HC4) return;
  int e = (int)(idx / HC4), q = (int)(idx - (size_t)e * HC4);
  int j = q << 2;
  int h = j / Cc;
  int s_, d_; edge_endpoints(e, ei, E, s_, d_);
  float a = p[(size_t)e * H + h];
  const float4 v = *(const float4*)(xl + (size_t)s_ * HC + j);
  float* dp = acc + (size_t)d_ * HC + j;
  atomicAdd(dp + 0, a * v.x);
  atomicAdd(dp + 1, a * v.y);
  atomicAdd(dp + 2, a * v.z);
  atomicAdd(dp + 3, a * v.w);
}

// h16 = f16(elu(acc))
__global__ __launch_bounds__(256)
void elu_cvt(const float* __restrict__ acc, _Float16* __restrict__ h16, size_t n) {
  size_t i = (size_t)blockIdx.x * blockDim.x + threadIdx.x;
  if (i < n) {
    float v = acc[i];
    v = v > 0.f ? v : expm1f(v);
    h16[i] = (_Float16)v;
  }
}

// final: write logits and log_softmax over 64 columns (one wave per row)
__global__ __launch_bounds__(256)
void logsoftmax64(const float* __restrict__ logits, float* __restrict__ outA,
                  float* __restrict__ outB, int Nn) {
  int wave = threadIdx.x >> 5, lane = threadIdx.x & 31;
  int row = blockIdx.x * 8 + wave;
  if (row >= Nn) return;
  const float* r = logits + (size_t)row * 64;
  float a0 = r[lane], a1 = r[lane + 32];
  float mx = fmaxf(a0, a1);
  for (int o = 16; o; o >>= 1) mx = fmaxf(mx, __shfl_xor(mx, o, 32));
  float se = __expf(a0 - mx) + __expf(a1 - mx);
  for (int o = 16; o; o >>= 1) se += __shfl_xor(se, o, 32);
  float lse = mx + logf(se);
  float* oa = outA + (size_t)row * 64;
  float* ob = outB + (size_t)row * 64;
  oa[lane] = a0;        oa[lane + 32] = a1;
  ob[lane] = a0 - lse;  ob[lane + 32] = a1 - lse;
}

// ---------------------------------------------------------------------------
// launcher
// ---------------------------------------------------------------------------
static inline unsigned blocks_for(size_t n, int bs) {
  return (unsigned)((n + (size_t)bs - 1) / bs);
}

extern "C" void kernel_launch(void* const* d_in, const int* in_sizes, int n_in,
                              void* d_out, int out_size, void* d_ws, size_t ws_size,
                              hipStream_t stream) {
  (void)n_in; (void)ws_size; (void)out_size;
  const float* x  = (const float*)d_in[0];
  const int*   ei = (const int*)d_in[1];
  const int N = in_sizes[0] / 512;
  const int E = in_sizes[1] / 2;
  const int Ntot = E + N;

  const float* Wl[3] = { (const float*)d_in[2],  (const float*)d_in[8],  (const float*)d_in[14] };
  const float* bl[3] = { (const float*)d_in[3],  (const float*)d_in[9],  (const float*)d_in[15] };
  const float* Wr[3] = { (const float*)d_in[4],  (const float*)d_in[10], (const float*)d_in[16] };
  const float* br[3] = { (const float*)d_in[5],  (const float*)d_in[11], (const float*)d_in[17] };
  const float* at[3] = { (const float*)d_in[6],  (const float*)d_in[12], (const float*)d_in[18] };
  const float* bo[3] = { (const float*)d_in[7],  (const float*)d_in[13], (const float*)d_in[19] };
  const int Hs[3] = { 8, 8, 1 };
  const int Cs[3] = { 32, 32, 64 };
  const int Ks[3] = { 512, 256, 256 };
  const int Ws[3] = { 512 * 256, 256 * 256, 256 * 64 };   // weight element counts

  // workspace layout
  char* ws = (char*)d_ws; size_t off = 0;
  auto alloc = [&](size_t bytes) -> char* {
    char* ptr = ws + off;
    off = (off + bytes + 255) & ~(size_t)255;
    return ptr;
  };
  _Float16* x16  = (_Float16*)alloc((size_t)N * 512 * sizeof(_Float16));
  _Float16* h16  = (_Float16*)alloc((size_t)N * 256 * sizeof(_Float16));
  _Float16* w16l[3]; _Float16* w16r[3];
  for (int L = 0; L < 3; ++L) {
    w16l[L] = (_Float16*)alloc((size_t)Ws[L] * sizeof(_Float16));
    w16r[L] = (_Float16*)alloc((size_t)Ws[L] * sizeof(_Float16));
  }
  float* xl   = (float*)alloc((size_t)N * 256 * sizeof(float));
  float* xr   = (float*)alloc((size_t)N * 256 * sizeof(float));
  float* acc  = (float*)alloc((size_t)N * 256 * sizeof(float));
  float* pbuf = (float*)alloc((size_t)Ntot * 8 * sizeof(float));
  float* mbuf = (float*)alloc((size_t)N * 8 * sizeof(float));
  float* sbuf = (float*)alloc((size_t)N * 8 * sizeof(float));

  // convert activations; pack weights into WMMA fragment order
  {
    size_t nx = (size_t)N * 512;
    cvt_f32_f16<<<blocks_for(nx, 256), 256, 0, stream>>>(x, x16, nx);
    for (int L = 0; L < 3; ++L) {
      const int K = Ks[L], M = Hs[L] * Cs[L];
      pack_w16<<<blocks_for((size_t)Ws[L], 256), 256, 0, stream>>>(Wl[L], w16l[L], K, M);
      pack_w16<<<blocks_for((size_t)Ws[L], 256), 256, 0, stream>>>(Wr[L], w16r[L], K, M);
    }
  }

  const unsigned rowTiles = (unsigned)((N + 15) / 16);

  for (int L = 0; L < 3; ++L) {
    const int H = Hs[L], Cc = Cs[L], K = Ks[L];
    const int HC = H * Cc;                       // 256, 256, 64
    const _Float16* A16 = (L == 0) ? x16 : h16;

    // dense transforms: xl = A*Wl + bl ; xr = A*Wr + br
    dim3 gg(rowTiles, (unsigned)(HC / 64));
    gemm_wmma_bias<<<gg, 128, 0, stream>>>(A16, w16l[L], bl[L], xl, N, K, HC);
    gemm_wmma_bias<<<gg, 128, 0, stream>>>(A16, w16r[L], br[L], xr, N, K, HC);

    // init acc = bias_out broadcast, m = -inf, s = 0
    init_layer<<<blocks_for((size_t)N * HC, 256), 256, 0, stream>>>(
        mbuf, sbuf, acc, bo[L], N, H, HC);

    const size_t totEH = (size_t)Ntot * H;
    edge_scores<<<blocks_for(totEH, 256), 256, 0, stream>>>(
        xl, xr, ei, at[L], pbuf, mbuf, E, Ntot, H, Cc);
    edge_exp<<<blocks_for(totEH, 256), 256, 0, stream>>>(
        pbuf, mbuf, sbuf, ei, E, Ntot, H);
    alpha_norm<<<blocks_for(totEH, 256), 256, 0, stream>>>(
        pbuf, sbuf, ei, E, Ntot, H);

    const size_t totAgg = (size_t)Ntot * (HC >> 2);
    aggregate4<<<blocks_for(totAgg, 256), 256, 0, stream>>>(
        pbuf, xl, ei, acc, E, Ntot, H, Cc);

    if (L < 2) {
      elu_cvt<<<blocks_for((size_t)N * HC, 256), 256, 0, stream>>>(
          acc, h16, (size_t)N * HC);
    } else {
      float* outA = (float*)d_out;
      float* outB = outA + (size_t)N * 64;
      logsoftmax64<<<blocks_for((size_t)N, 8), 256, 0, stream>>>(acc, outA, outB, N);
    }
  }
}
